// PatternEncoder_61151744360779
// MI455X (gfx1250) — 8-bit matrix operand order repaired and checked by reading, compiled, not run on hardware
//
#include <hip/hip_runtime.h>
#include <hip/hip_bf16.h>
#include <stdint.h>

#define NN 4096
#define DD 32
#define EE 32768
#define BB 8

#define FP8_ONE 0x38

typedef int   v16i __attribute__((ext_vector_type(16)));
typedef float v8f  __attribute__((ext_vector_type(8)));

#define A_BYTES ((size_t)NN * (size_t)NN)
#define T_BYTES ((size_t)NN * 4)

__global__ void clear_ws_kernel(uint4* ws, size_t n16) {
  size_t i = (size_t)blockIdx.x * blockDim.x + threadIdx.x;
  const size_t stride = (size_t)gridDim.x * blockDim.x;
  const uint4 z = {0u, 0u, 0u, 0u};
  for (; i < n16; i += stride) ws[i] = z;
}

__global__ void build_adj_kernel(const int* __restrict__ ei,
                                 const int* __restrict__ batch,
                                 uint8_t* __restrict__ A) {
  int e = blockIdx.x * blockDim.x + threadIdx.x;
  if (e >= EE) return;
  int s = ei[e];
  int d = ei[EE + e];
  if (s != d && batch[s] == batch[d]) {
    A[(size_t)s * NN + d] = FP8_ONE;
    A[(size_t)d * NN + s] = FP8_ONE;
  }
}

__device__ __forceinline__ v16i load_a_frag(const uint8_t* __restrict__ arow,
                                            int k, int hi) {
  v16i a;
#pragma unroll
  for (int h2 = 0; h2 < 2; ++h2) {
#pragma unroll
    for (int p = 0; p < 4; ++p) {
      const uint2 v = *(const uint2*)(arow + k + h2 * 64 + (p >> 1) * 32 + hi * 16 + (p & 1) * 8);
      a[h2 * 8 + 2 * p + 0] = (int)v.x;
      a[h2 * 8 + 2 * p + 1] = (int)v.y;
    }
  }
  return a;
}

__device__ __forceinline__ v16i load_b_frag(const uint8_t* __restrict__ brow,
                                            int k, int hi) {
  v16i b;
#pragma unroll
  for (int g = 0; g < 4; ++g) {
    const int4 v = *(const int4*)(brow + k + g * 32 + hi * 16);
    b[4 * g + 0] = v.x;
    b[4 * g + 1] = v.y;
    b[4 * g + 2] = v.z;
    b[4 * g + 3] = v.w;
  }
  return b;
}

__global__ __launch_bounds__(256) void tri_count_kernel(
    const uint8_t* __restrict__ A, float* __restrict__ t) {
  const int lane = threadIdx.x & 31;
  const int wave = threadIdx.x >> 5;
  const int colPair = blockIdx.x * 8 + wave;
  const int rowTile = blockIdx.y;
  const int rowBase = rowTile * 16;
  const int colBase0 = colPair * 32;
  const int colBase1 = colBase0 + 16;
  const int hi = lane >> 4;
  const int ml = lane & 15;

  const uint8_t* arow  = A + (size_t)(rowBase + ml) * NN;
  const uint8_t* brow0 = A + (size_t)(colBase0 + ml) * NN;
  const uint8_t* brow1 = A + (size_t)(colBase1 + ml) * NN;

  v8f c0 = {};
  v8f c1 = {};

  v16i a_cur  = load_a_frag(arow, 0, hi);
  v16i b0_cur = load_b_frag(brow0, 0, hi);
  v16i b1_cur = load_b_frag(brow1, 0, hi);

  for (int k = 0; k < NN; k += 128) {
    const int kn = (k + 128) & (NN - 1);
    __builtin_prefetch(arow + k + 256, 0, 1);
    __builtin_prefetch(brow0 + k + 256, 0, 1);
    __builtin_prefetch(brow1 + k + 256, 0, 1);
    v16i a_nxt  = load_a_frag(arow, kn, hi);
    v16i b0_nxt = load_b_frag(brow0, kn, hi);
    v16i b1_nxt = load_b_frag(brow1, kn, hi);

    c0 = __builtin_amdgcn_wmma_f32_16x16x128_fp8_fp8(a_cur, b0_cur, (short)0,
                                                     c0, false, false);
    c1 = __builtin_amdgcn_wmma_f32_16x16x128_fp8_fp8(a_cur, b1_cur, (short)0,
                                                     c1, false, false);
    a_cur = a_nxt;
    b0_cur = b0_nxt;
    b1_cur = b1_nxt;
  }

#pragma unroll
  for (int r = 0; r < 8; ++r) {
    const int row = rowBase + r + hi * 8;
    const size_t rb = (size_t)row * NN;
    float v = (A[rb + colBase0 + ml] != 0 ? c0[r] : 0.0f) +
              (A[rb + colBase1 + ml] != 0 ? c1[r] : 0.0f);
#pragma unroll
    for (int off = 8; off >= 1; off >>= 1) v += __shfl_xor(v, off, 16);
    if (ml == 0) atomicAdd(&t[row], 0.5f * v);
  }
}

__global__ __launch_bounds__(256) void pool_linear_kernel(
    const float* __restrict__ x, const int* __restrict__ batch,
    const float* __restrict__ t, const float* __restrict__ W,
    const float* __restrict__ bias, float* __restrict__ out) {
  __shared__ float s_tri[BB][DD];
  __shared__ float s_non[BB][DD];
  __shared__ float s_tc[BB];
  __shared__ float s_nc[BB];
  __shared__ float s_add[BB][DD];
  __shared__ float s_mean[BB][DD];

  const int tid = threadIdx.x;
  for (int i = tid; i < BB * DD; i += 256) {
    s_tri[i / DD][i % DD] = 0.0f;
    s_non[i / DD][i % DD] = 0.0f;
  }
  if (tid < BB) { s_tc[tid] = 0.0f; s_nc[tid] = 0.0f; }
  __syncthreads();

  for (int u = tid; u < NN; u += 256) {
    const int b = batch[u];
    const float tu = t[u];
    if (tu != 0.0f) {
      atomicAdd(&s_tc[b], tu);
      for (int d = 0; d < DD; ++d)
        atomicAdd(&s_tri[b][d], tu * x[u * DD + d]);
    } else {
      atomicAdd(&s_nc[b], 1.0f);
      for (int d = 0; d < DD; ++d)
        atomicAdd(&s_non[b][d], x[u * DD + d]);
    }
  }
  __syncthreads();

  for (int i = tid; i < BB * DD; i += 256) {
    const int b = i / DD, d = i % DD;
    const float add = s_tri[b][d] * (1.0f / 3.0f) + s_non[b][d];
    float cnt = s_tc[b] * (1.0f / 3.0f) + s_nc[b];
    cnt = fmaxf(cnt, 1.0f);
    s_add[b][d] = add;
    s_mean[b][d] = add / cnt;
  }
  __syncthreads();

  for (int i = tid; i < BB * DD; i += 256) {
    const int b = i / DD, d = i % DD;
    float acc = bias[d];
    for (int j = 0; j < DD; ++j) {
      acc += s_mean[b][j] * W[j * DD + d];
      acc += s_add[b][j] * W[(DD + j) * DD + d];
    }
    out[i] = acc;
  }
}

extern "C" void kernel_launch(void* const* d_in, const int* in_sizes, int n_in,
                              void* d_out, int out_size, void* d_ws,
                              size_t ws_size, hipStream_t stream) {
  (void)in_sizes; (void)n_in; (void)out_size; (void)ws_size;
  const float* x     = (const float*)d_in[0];
  const int*   ei    = (const int*)d_in[1];
  const int*   batch = (const int*)d_in[2];
  const float* W     = (const float*)d_in[3];
  const float* bias  = (const float*)d_in[4];

  uint8_t* A = (uint8_t*)d_ws;
  float*   t = (float*)((char*)d_ws + A_BYTES);

  const size_t n16 = (A_BYTES + T_BYTES) / 16;
  clear_ws_kernel<<<4096, 256, 0, stream>>>((uint4*)d_ws, n16);

  build_adj_kernel<<<EE / 256, 256, 0, stream>>>(ei, batch, A);

  dim3 grid(NN / 16 / 8 / 2, NN / 16);
  tri_count_kernel<<<grid, 256, 0, stream>>>(A, t);

  pool_linear_kernel<<<1, 256, 0, stream>>>(x, batch, t, W, bias,
                                            (float*)d_out);
}
